// MatrixDFTLayer_11879879542476
// MI455X (gfx1250) — matrix operand order repaired and checked by reading, not compiled, not run on hardware
//
#include <hip/hip_runtime.h>

// ---------------------------------------------------------------------------
// DFT-as-GEMM for MI455X (gfx1250):  yr = x @ Wr^T, yi = x @ Wi^T
//   x : [16384, 1024] f32,  Wr/Wi : [1024, 1024] f32 (row-major, W[k][n])
//   out[b][k] = (yr, yi) interleaved -> [16384, 1024, 2] f32
// f16 inputs + f32 accumulate via v_wmma_f32_16x16x32_f16 (wave32).
// ---------------------------------------------------------------------------

typedef __attribute__((ext_vector_type(4)))  _Float16 v4h;
typedef __attribute__((ext_vector_type(8)))  _Float16 v8h;
typedef __attribute__((ext_vector_type(16))) _Float16 v16h;
typedef __attribute__((ext_vector_type(4)))  float    v4f;
typedef __attribute__((ext_vector_type(8)))  float    v8f;

#define BATCH 16384
#define NN    1024
#define KK    1024

#define M_BLK 64
#define N_BLK 128
#define K_BLK 32

__device__ __forceinline__ v16h frag_combine(v8h lo, v8h hi) {
    v16h r;
#pragma unroll
    for (int i = 0; i < 8; ++i) { r[i] = lo[i]; r[i + 8] = hi[i]; }
    return r;
}

__global__ __launch_bounds__(256) void MatrixDFTLayer_wmma_kernel(
    const float* __restrict__ x,
    const float* __restrict__ wr,
    const float* __restrict__ wi,
    float* __restrict__ out)
{
    // LDS staging tiles, f16 (converted from f32 on the fly).
    __shared__ __align__(16) _Float16 Ash [M_BLK][K_BLK];   //  4 KB : x tile   [m][k]
    __shared__ __align__(16) _Float16 Brsh[N_BLK][K_BLK];   //  8 KB : Wr tile  [n][k]
    __shared__ __align__(16) _Float16 Bish[N_BLK][K_BLK];   //  8 KB : Wi tile  [n][k]

    const int tid  = threadIdx.x;
    const int lane = tid & 31;
    const int wv   = tid >> 5;      // wave id 0..7
    const int wM   = wv >> 2;       // 0..1  (M direction)
    const int wN   = wv & 3;        // 0..3  (N direction)
    const int hsel = lane >> 4;     // lane half: 0 (lanes 0-15) / 1 (lanes 16-31)
    const int lrow = lane & 15;

    const int b0 = blockIdx.y * M_BLK;   // batch-row origin of block tile
    const int n0 = blockIdx.x * N_BLK;   // output-freq origin of block tile

    v8f acc_r[2][2];
    v8f acc_i[2][2];
#pragma unroll
    for (int ms = 0; ms < 2; ++ms)
#pragma unroll
        for (int ns = 0; ns < 2; ++ns) {
            acc_r[ms][ns] = (v8f)0.0f;
            acc_i[ms][ns] = (v8f)0.0f;
        }

    for (int p0 = 0; p0 < KK; p0 += K_BLK) {
        __syncthreads();  // previous iteration's fragment reads complete

        // ---- stage A (x) tile: 64x32 f32 -> f16, 2 x float4 per thread ----
#pragma unroll
        for (int i = 0; i < 2; ++i) {
            const int f   = tid + i * 256;        // float4 slot 0..511
            const int row = f >> 3;               // 0..63
            const int col = (f & 7) << 2;         // 0,4,...,28
            v4f v = *(const v4f*)(x + (size_t)(b0 + row) * KK + p0 + col);
            v4h h;
            h[0] = (_Float16)v[0]; h[1] = (_Float16)v[1];
            h[2] = (_Float16)v[2]; h[3] = (_Float16)v[3];
            *(v4h*)&Ash[row][col] = h;            // ds_store_b64
        }

        // ---- stage B tiles (Wr, Wi): 128x32 f32 -> f16, 4 x float4 each ----
#pragma unroll
        for (int i = 0; i < 4; ++i) {
            const int f   = tid + i * 256;        // float4 slot 0..1023
            const int row = f >> 3;               // 0..127  (output index k)
            const int col = (f & 7) << 2;         // K offset
            const size_t g = (size_t)(n0 + row) * KK + p0 + col;

            v4f vr = *(const v4f*)(wr + g);
            v4h hr;
            hr[0] = (_Float16)vr[0]; hr[1] = (_Float16)vr[1];
            hr[2] = (_Float16)vr[2]; hr[3] = (_Float16)vr[3];
            *(v4h*)&Brsh[row][col] = hr;

            v4f vi = *(const v4f*)(wi + g);
            v4h hi;
            hi[0] = (_Float16)vi[0]; hi[1] = (_Float16)vi[1];
            hi[2] = (_Float16)vi[2]; hi[3] = (_Float16)vi[3];
            *(v4h*)&Bish[row][col] = hi;
        }

        __syncthreads();  // tiles visible to all waves

        // ---- load fragments in contiguous k order (same order for A and B) ----
        // A 16x32 f16: lane<16 -> M=lane,   K = 0..15
        //              lane>=16 -> M=lane-16, K = 16..31
        v16h afrag[2];
#pragma unroll
        for (int ms = 0; ms < 2; ++ms) {
            const int m = wM * 32 + ms * 16 + lrow;
            v8h lo = *(const v8h*)&Ash[m][hsel * 16];       // ds_load_b128
            v8h hi = *(const v8h*)&Ash[m][hsel * 16 + 8];   // ds_load_b128
            afrag[ms] = frag_combine(lo, hi);
        }

        // B 32x16 f16: lane<16 -> N=lane,   K = 0..15 ; lane>=16 -> N=lane-16, K = 16..31
        // Bsh is stored [n][k], so each lane's 16 halves are 32 contiguous bytes.
        v16h brfrag[2], bifrag[2];
#pragma unroll
        for (int ns = 0; ns < 2; ++ns) {
            const int j = wN * 32 + ns * 16 + lrow;
            v8h rlo = *(const v8h*)&Brsh[j][hsel * 16];
            v8h rhi = *(const v8h*)&Brsh[j][hsel * 16 + 8];
            brfrag[ns] = frag_combine(rlo, rhi);
            v8h ilo = *(const v8h*)&Bish[j][hsel * 16];
            v8h ihi = *(const v8h*)&Bish[j][hsel * 16 + 8];
            bifrag[ns] = frag_combine(ilo, ihi);
        }

        // ---- 8 WMMAs per wave per K-step ----
#pragma unroll
        for (int ms = 0; ms < 2; ++ms) {
#pragma unroll
            for (int ns = 0; ns < 2; ++ns) {
                acc_r[ms][ns] = __builtin_amdgcn_wmma_f32_16x16x32_f16(
                    false, afrag[ms], false, brfrag[ns],
                    (short)0, acc_r[ms][ns], false, false);
                acc_i[ms][ns] = __builtin_amdgcn_wmma_f32_16x16x32_f16(
                    false, afrag[ms], false, bifrag[ns],
                    (short)0, acc_i[ms][ns], false, false);
            }
        }
    }

    // ---- epilogue: interleave (yr, yi) and store as float2 (global_store_b64) ----
    // C/D layout: VGPR v -> M = v + 8*hsel (within 16-row subtile), N = lrow.
#pragma unroll
    for (int ms = 0; ms < 2; ++ms) {
#pragma unroll
        for (int ns = 0; ns < 2; ++ns) {
            const int kcol = n0 + wN * 32 + ns * 16 + lrow;
#pragma unroll
            for (int v = 0; v < 8; ++v) {
                const int m = b0 + wM * 32 + ms * 16 + hsel * 8 + v;
                float2 val;
                val.x = acc_r[ms][ns][v];
                val.y = acc_i[ms][ns][v];
                *(float2*)(out + ((size_t)m * NN + kcol) * 2) = val;
            }
        }
    }
}

extern "C" void kernel_launch(void* const* d_in, const int* in_sizes, int n_in,
                              void* d_out, int out_size, void* d_ws, size_t ws_size,
                              hipStream_t stream) {
    const float* x  = (const float*)d_in[0];   // [16384, 1024]
    const float* wr = (const float*)d_in[1];   // [1024, 1024]
    const float* wi = (const float*)d_in[2];   // [1024, 1024]
    float* out = (float*)d_out;                // [16384, 1024, 2]

    dim3 grid(NN / N_BLK, BATCH / M_BLK);      // (8, 256)
    dim3 block(256);
    MatrixDFTLayer_wmma_kernel<<<grid, block, 0, stream>>>(x, wr, wi, out);
}
